// GeneRouter_22969485099582
// MI455X (gfx1250) — matrix operand order repaired and checked by reading, not compiled, not run on hardware
//
#include <hip/hip_runtime.h>
#include <hip/hip_bf16.h>

typedef __bf16 bf16_t;
typedef __bf16 v16bf __attribute__((ext_vector_type(16)));
typedef __bf16 v8bf  __attribute__((ext_vector_type(8)));
typedef __bf16 v4bf  __attribute__((ext_vector_type(4)));
typedef float  v8f   __attribute__((ext_vector_type(8)));

#define C_   16
#define L_   8192
#define S_   8191               // L-1
#define NT_  (C_ * S_)          // 131056 tokens
#define E_   512
#define H_   256
#define M_   128
#define KSEL 30

// packed bf16 weight fragments in d_ws: [kt][nt][lane][16 bf16] (B-operand layout)
#define PW1_ELEMS (16 * 16 * 32 * 16)   // 131072  (E/32 x H/16)
#define PW2_ELEMS (8 * 8 * 32 * 16)     // 32768   (H/32 x M/16)

// Use the gfx1250 async global->LDS copy path (ASYNCcnt-tracked).
// Flip to 0 if the assembler rejects the mnemonic.
#define USE_ASYNC_COPY 1

// ---------------------------------------------------------------------------
// Pack W1/W2 (fp32 row-major) into per-lane WMMA-B fragment order, bf16.
// B 32x16 bf16 layout: lane&15 -> column; lane>>4 selects K sub-offset 8;
// lane holds K = 8h+j (j<8) and 16+8h+(j-8) (j>=8), matching the A layout.
// ---------------------------------------------------------------------------
__global__ __launch_bounds__(256) void pack_weights(const float* __restrict__ W1,
                                                    const float* __restrict__ W2,
                                                    bf16_t* __restrict__ ws) {
    int i = blockIdx.x * 256 + threadIdx.x;
    if (i < PW1_ELEMS) {
        int j    = i & 15;
        int lane = (i >> 4) & 31;
        int nt   = (i >> 9) & 15;
        int kt   = i >> 13;
        int col  = nt * 16 + (lane & 15);
        int k    = kt * 32 + (lane >> 4) * 8 + (j & 7) + (j >> 3) * 16;
        ws[i] = (bf16_t)W1[k * H_ + col];
    } else if (i < PW1_ELEMS + PW2_ELEMS) {
        int p    = i - PW1_ELEMS;
        int j    = p & 15;
        int lane = (p >> 4) & 31;
        int nt   = (p >> 9) & 7;
        int kt   = p >> 12;
        int col  = nt * 16 + (lane & 15);
        int k    = kt * 32 + (lane >> 4) * 8 + (j & 7) + (j >> 3) * 16;
        ws[PW1_ELEMS + p] = (bf16_t)W2[k * M_ + col];
    }
}

// ---------------------------------------------------------------------------
// Stage 64KB of packed bf16 fragments global -> LDS (128 threads, 512B each).
// Async path: global_load_async_to_lds_b128 + s_wait_asynccnt (gfx1250 TDM-ish
// async pipe, no VGPR round trip).  Caller must __syncthreads() afterwards.
// ---------------------------------------------------------------------------
__device__ __forceinline__ void stage64k(const bf16_t* __restrict__ src,
                                         bf16_t* dst, int tid) {
    const bf16_t* s = src + tid * 256;   // 256 bf16 = 512 B per thread
    bf16_t*       d = dst + tid * 256;
#if USE_ASYNC_COPY
    unsigned ldsoff = (unsigned)(size_t)d;   // flat->LDS: low 32 bits = LDS offset
    #pragma unroll
    for (int i = 0; i < 32; ++i) {
        asm volatile("global_load_async_to_lds_b128 %0, %1, off"
                     :: "v"(ldsoff + i * 16), "v"(s + i * 8)
                     : "memory");
    }
    asm volatile("s_wait_asynccnt 0" ::: "memory");
#else
    #pragma unroll
    for (int i = 0; i < 32; ++i)
        *(uint4*)(d + i * 8) = *(const uint4*)(s + i * 8);
#endif
}

// ---------------------------------------------------------------------------
// Fused: h = relu(x@W1+b1); logits = h@W2+b2; gumbel-softmax; top-30 smooth
// mask; padding mask.  64 tokens/block, 4 waves, 16 rows/wave, bf16 WMMA.
// Weights are staged once per block into LDS and shared by all 4 waves.
// ---------------------------------------------------------------------------
__global__ __launch_bounds__(128) void gene_router_fused(
    const float* __restrict__ x,
    const int*   __restrict__ kpm,
    const float* __restrict__ noise,
    const float* __restrict__ b1,
    const float* __restrict__ b2,
    const bf16_t* __restrict__ pw,
    float* __restrict__ out)     // [NT*M] out, then [NT] kpm-as-float
{
    __shared__ __align__(16) bf16_t sX[64][136];   // x K-chunk, bf16 (+8 pad)
    __shared__ __align__(16) bf16_t sH[64][264];   // hidden, bf16 (+8 pad)
    __shared__ __align__(16) bf16_t sB[32768];     // 64KB staged weight frags

    const int tid  = threadIdx.x;
    const int lane = tid & 31;
    const int w    = tid >> 5;
    const int base = blockIdx.x * 64;
    const bf16_t* pW1 = pw;
    const bf16_t* pW2 = pw + PW1_ELEMS;

    // second tuple output: kpm (bool) as float
    if (tid < 64) {
        int tok = base + tid;
        if (tok < NT_) {
            unsigned c = (unsigned)tok / S_;
            out[(size_t)NT_ * M_ + tok] = kpm[tok + c + 1] ? 1.0f : 0.0f;
        }
    }

    const int arow  = w * 16 + (lane & 15);  // A-fragment row (tile-local)
    const int khalf = (lane >> 4) * 8;       // A layout K sub-offset

    // ----------------- GEMM1: 16x256 per wave, K=512 in 4 LDS chunks -------
    v8f acc1[16] = {};
    for (int chunk = 0; chunk < 4; ++chunk) {
        __syncthreads();      // previous iteration's sX/sB readers done
        {   // cooperative fp32 load -> bf16 LDS stage (2 threads per row)
            int r       = tid >> 1;
            int halfsel = (tid & 1) * 64;
            int tok = base + r; if (tok >= NT_) tok = NT_ - 1;
            unsigned c = (unsigned)tok / S_;
            const float* src = x + (size_t)(tok + c + 1) * E_ + chunk * 128 + halfsel;
            #pragma unroll
            for (int i = 0; i < 16; ++i) {
                float4 f = *(const float4*)(src + i * 4);
                v4bf bv;
                bv[0] = (bf16_t)f.x; bv[1] = (bf16_t)f.y;
                bv[2] = (bf16_t)f.z; bv[3] = (bf16_t)f.w;
                *(v4bf*)&sX[r][halfsel + i * 4] = bv;
            }
        }
        // stage this K-chunk of packed W1 (4 kt x 16 nt x 1KB = 64KB)
        stage64k(pW1 + chunk * 32768, sB, tid);
        __syncthreads();

        #pragma unroll
        for (int kt = 0; kt < 4; ++kt) {
            int c0 = kt * 32 + khalf;
            v8bf a0 = *(const v8bf*)&sX[arow][c0];
            v8bf a1 = *(const v8bf*)&sX[arow][c0 + 16];
            v16bf a;
            #pragma unroll
            for (int i = 0; i < 8; ++i) { a[i] = a0[i]; a[i + 8] = a1[i]; }
            #pragma unroll
            for (int nt = 0; nt < 16; ++nt) {
                v16bf b = *(const v16bf*)&sB[(((kt * 16 + nt) * 32 + lane) << 4)];
                acc1[nt] = __builtin_amdgcn_wmma_f32_16x16x32_bf16(
                    false, a, false, b, (short)0, acc1[nt], false, false);
            }
        }
    }

    // bias + ReLU -> sH (bf16, row-major)
    {
        const int colbase = lane & 15;
        const int rhalf   = (lane >> 4) * 8;
        #pragma unroll
        for (int nt = 0; nt < 16; ++nt) {
            float bb = b1[nt * 16 + colbase];
            #pragma unroll
            for (int v = 0; v < 8; ++v) {
                float h = fmaxf(acc1[nt][v] + bb, 0.0f);
                sH[w * 16 + rhalf + v][nt * 16 + colbase] = (bf16_t)h;
            }
        }
    }
    __syncthreads();                     // sH visible; GEMM1 sB readers done
    stage64k(pW2, sB, tid);              // all of packed W2 (64KB)
    __syncthreads();

    // ----------------- GEMM2: 16x128 per wave, K=256 -----------------------
    v8f acc2[8] = {};
    #pragma unroll
    for (int kt = 0; kt < 8; ++kt) {
        int c0 = kt * 32 + khalf;
        v8bf a0 = *(const v8bf*)&sH[arow][c0];
        v8bf a1 = *(const v8bf*)&sH[arow][c0 + 16];
        v16bf a;
        #pragma unroll
        for (int i = 0; i < 8; ++i) { a[i] = a0[i]; a[i + 8] = a1[i]; }
        #pragma unroll
        for (int nt = 0; nt < 8; ++nt) {
            v16bf b = *(const v16bf*)&sB[(((kt * 8 + nt) * 32 + lane) << 4)];
            acc2[nt] = __builtin_amdgcn_wmma_f32_16x16x32_bf16(
                false, a, false, b, (short)0, acc2[nt], false, false);
        }
    }

    // ----------------- Epilogue: gumbel softmax + top-30 smooth mask -------
    const int colbase = lane & 15;
    const int rhalf   = (lane >> 4) * 8;
    float b2c[8];
    #pragma unroll
    for (int nt = 0; nt < 8; ++nt) b2c[nt] = b2[nt * 16 + colbase];

    #pragma unroll
    for (int v = 0; v < 8; ++v) {
        int tok  = base + w * 16 + rhalf + v;
        int tokn = tok < NT_ ? tok : NT_ - 1;
        float pv[8];
        #pragma unroll
        for (int nt = 0; nt < 8; ++nt) {
            float u = noise[(size_t)tokn * M_ + nt * 16 + colbase];
            float g = -__logf(-__logf(u + 1e-20f) + 1e-20f);
            pv[nt] = acc2[nt][v] + b2c[nt] + g;          // TEMP = TAU = 1
        }
        // row max / sum across the owning 16-lane half (+8 local values)
        float mx = pv[0];
        #pragma unroll
        for (int nt = 1; nt < 8; ++nt) mx = fmaxf(mx, pv[nt]);
        #pragma unroll
        for (int m = 1; m <= 8; m <<= 1) mx = fmaxf(mx, __shfl_xor(mx, m, 32));
        float s = 0.0f;
        #pragma unroll
        for (int nt = 0; nt < 8; ++nt) { pv[nt] = __expf(pv[nt] - mx); s += pv[nt]; }
        #pragma unroll
        for (int m = 1; m <= 8; m <<= 1) s += __shfl_xor(s, m, 32);
        float inv = 1.0f / s;
        #pragma unroll
        for (int nt = 0; nt < 8; ++nt) pv[nt] *= inv;

        // binary search for the 30th-largest probability (probs in [0,1])
        float lo = 0.0f, hi = 1.0f;
        for (int it = 0; it < 24; ++it) {
            float mid = 0.5f * (lo + hi);
            float cnt = 0.0f;
            #pragma unroll
            for (int nt = 0; nt < 8; ++nt) cnt += (pv[nt] > mid) ? 1.0f : 0.0f;
            #pragma unroll
            for (int m = 1; m <= 8; m <<= 1) cnt += __shfl_xor(cnt, m, 32);
            if (cnt >= (float)KSEL) lo = mid; else hi = mid;
        }
        float thr = 0.5f * (lo + hi);

        if (tok < NT_) {
            unsigned c = (unsigned)tok / S_;
            float keep = kpm[tok + c + 1] ? 0.0f : 1.0f;
            float* o = out + (size_t)tok * M_;
            #pragma unroll
            for (int nt = 0; nt < 8; ++nt) {
                float p  = pv[nt];
                float sm = 1.0f / (1.0f + __expf((thr - p) * 100.0f)); // sigmoid((p-thr)/0.01)
                o[nt * 16 + colbase] = p * sm * keep;
            }
        }
    }
}

// ---------------------------------------------------------------------------
extern "C" void kernel_launch(void* const* d_in, const int* in_sizes, int n_in,
                              void* d_out, int out_size, void* d_ws, size_t ws_size,
                              hipStream_t stream) {
    (void)in_sizes; (void)n_in; (void)out_size; (void)ws_size;
    const float* x     = (const float*)d_in[0];
    const int*   kpmv  = (const int*)d_in[1];
    const float* noise = (const float*)d_in[2];
    const float* W1    = (const float*)d_in[3];
    const float* b1    = (const float*)d_in[4];
    const float* W2    = (const float*)d_in[5];
    const float* b2    = (const float*)d_in[6];
    bf16_t* pw  = (bf16_t*)d_ws;          // needs 327,680 B of workspace
    float*  out = (float*)d_out;

    const int packN = PW1_ELEMS + PW2_ELEMS;
    pack_weights<<<(packN + 255) / 256, 256, 0, stream>>>(W1, W2, pw);

    const int nblocks = (NT_ + 63) / 64;   // 2048
    gene_router_fused<<<nblocks, 128, 0, stream>>>(x, kpmv, noise, b1, b2, pw, out);
}
